// LSTM_FEAT_2_24524263260275
// MI455X (gfx1250) — matrix operand order repaired and checked by reading, not compiled, not run on hardware
//
#include <hip/hip_runtime.h>

// ---------------------------------------------------------------------------
// Persistent bf16-WMMA LSTM pipeline for gfx1250 (MI455X).
//  - Prep kernels convert all fp32 weights into bf16 tiles for the
//    the V_WMMA_F32_16X16X32_BF16 A operand, k = 16*half + slot as B reads it (one 32B v16bf / lane).
//  - Main kernel: 16 workgroups x 256 threads (8 waves), persistent over
//    T=4096 steps, global sense-counter barrier between stages.
//  - Matvec trick: B operand = vector chunk broadcast to all 16 columns,
//    result read from column 0 (lane 0 -> rows 0..7, lane 16 -> rows 8..15).
//  - Activation vectors staged into LDS with GLOBAL_LOAD_ASYNC_TO_LDS_B128
//    (ASYNCcnt) when the toolchain exposes the builtin.
// ---------------------------------------------------------------------------

#define T_STEPS 4096
#define NWG 16
#define TPB 256

typedef __attribute__((ext_vector_type(16))) __bf16 v16bf;
typedef __attribute__((ext_vector_type(8)))  float  v8f;
typedef __attribute__((ext_vector_type(4)))  int    i4vec;
typedef __attribute__((address_space(1))) i4vec* gp4_t;  // global int4*
typedef __attribute__((address_space(3))) i4vec* lp4_t;  // LDS int4*

#if __has_builtin(__builtin_amdgcn_global_load_async_to_lds_b128)
#define HAVE_ASYNC_LDS 1
#else
#define HAVE_ASYNC_LDS 0
#endif

// ---- workspace layout (byte offsets in d_ws) ----
static constexpr size_t OFF_CTRL = 0;        // unsigned[64] (barrier counter)
static constexpr size_t OFF_INP  = 4096;     // __bf16[640]  feat|embed
static constexpr size_t OFF_H1   = 8192;     // __bf16[2][1024] double buffer
static constexpr size_t OFF_H2   = 12288;    // __bf16[2][1024] double buffer
static constexpr size_t OFF_C1   = 16384;    // float[1024]
static constexpr size_t OFF_C2   = 20480;    // float[1024]
static constexpr size_t OFF_Y    = 24576;    // float[512]

static constexpr size_t tileBytes(int rows, int kc) {
  return (size_t)(rows / 16) * (size_t)kc * 1024;  // 16x32 bf16 tile = 1KB
}
static constexpr size_t OFF_WS_T   = 32768;
static constexpr size_t OFF_WIH1_T = OFF_WS_T   + tileBytes(512,  12);
static constexpr size_t OFF_WHH1_T = OFF_WIH1_T + tileBytes(4096, 20);
static constexpr size_t OFF_WIH2_T = OFF_WHH1_T + tileBytes(4096, 32);
static constexpr size_t OFF_WHH2_T = OFF_WIH2_T + tileBytes(4096, 32);
static constexpr size_t OFF_WL_T   = OFF_WHH2_T + tileBytes(4096, 32);
static constexpr size_t OFF_WM_T   = OFF_WL_T   + tileBytes(512,  32);
static constexpr size_t WS_NEEDED  = OFF_WM_T   + tileBytes(128,  16);

// ---------------------------------------------------------------------------
// Prep: fp32 row-major W[rows x cols] -> bf16 tiles in WMMA A layout.
// Contiguous k order, matched by B (not ISA table order): lane l (half = l>>4, m = l&15) holds, in
// v16bf slot s:  K = 16*half + s,  M = m.
// Tile storage: [tileR*KC + kc][lane][slot] contiguous (32B per lane).
// ---------------------------------------------------------------------------
__global__ void tile_weights_k(const float* __restrict__ W, __bf16* __restrict__ Wt,
                               int rows, int cols, int KC) {
  size_t total = (size_t)(rows / 16) * (size_t)KC * 512;
  for (size_t i = (size_t)blockIdx.x * blockDim.x + threadIdx.x; i < total;
       i += (size_t)gridDim.x * blockDim.x) {
    int slot = (int)(i & 15);
    int lane = (int)((i >> 4) & 31);
    size_t tile = i >> 9;
    int kc = (int)(tile % (size_t)KC);
    int tR = (int)(tile / (size_t)KC);
    int v = slot >> 1, e = slot & 1;
    int kof = ((lane >> 4) << 4) + slot; (void)v; (void)e;
    int k = kc * 32 + kof;
    int r = tR * 16 + (lane & 15);
    float val = (r < rows && k < cols) ? W[(size_t)r * cols + k] : 0.0f;
    Wt[i] = (__bf16)val;
  }
}

__global__ void init_state_k(char* ws) {
  int i = blockIdx.x * blockDim.x + threadIdx.x;
  if (i < 64) ((unsigned*)(ws + OFF_CTRL))[i] = 0u;
  if (i < 1024) {
    ((float*)(ws + OFF_C1))[i] = 0.0f;
    ((float*)(ws + OFF_C2))[i] = 0.0f;
  }
  if (i < 2048) {
    ((__bf16*)(ws + OFF_H1))[i] = (__bf16)0.0f;
    ((__bf16*)(ws + OFF_H2))[i] = (__bf16)0.0f;
  }
  if (i < 640) ((__bf16*)(ws + OFF_INP))[i] = (__bf16)0.0f;
}

// ---------------------------------------------------------------------------
// Device helpers
// ---------------------------------------------------------------------------
__device__ __forceinline__ float sigmoidf_(float x) { return 1.0f / (1.0f + __expf(-x)); }

// Async DMA 16B global -> LDS (ASYNCcnt tracked), if toolchain exposes it.
__device__ __forceinline__ void async_cp16(void* g, void* l) {
#if HAVE_ASYNC_LDS
  __builtin_amdgcn_global_load_async_to_lds_b128((gp4_t)g, (lp4_t)l, 0, 0);
#else
  (void)g; (void)l;
#endif
}

__device__ __forceinline__ void wait_async0() {
#if HAVE_ASYNC_LDS
#if __has_builtin(__builtin_amdgcn_s_wait_asynccnt)
  __builtin_amdgcn_s_wait_asynccnt(0);
#else
  asm volatile("s_wait_asynccnt 0x0" ::: "memory");
#endif
#endif
}

// Accumulate one 16-row tile of W @ vec:   acc += A(16x32 chunks) x Bcast(vec)
// vec: bf16, length KC*32 (LDS). B columns all identical.
__device__ __forceinline__ v8f mv_tile(const v16bf* __restrict__ A, int tileR, int KC,
                                       const __bf16* __restrict__ vec, int lane, v8f acc) {
  const int hs = (lane >> 4) << 4;  // half-wave selects K 0..15 vs 16..31
  for (int kc = 0; kc < KC; ++kc) {
    const v16bf* ap = A + ((size_t)tileR * KC + kc) * 32 + lane;
    if (kc + 1 < KC) __builtin_prefetch((const void*)(ap + 32), 0, 3); // global_prefetch_b8 (near)
    v16bf a = *ap;
    v16bf b = *(const v16bf*)(vec + kc * 32 + hs);
    acc = __builtin_amdgcn_wmma_f32_16x16x32_bf16(false, a, false, b, (short)0, acc,
                                                  false, false);
  }
  return acc;
}

// Global barrier across the persistent NWG workgroups (monotone counter).
__device__ __forceinline__ void gbarrier(unsigned* cnt, unsigned& target) {
  __syncthreads();
  __threadfence();
  target += NWG;
  if (threadIdx.x == 0) {
    atomicAdd(cnt, 1u);
    while (__hip_atomic_load(cnt, __ATOMIC_ACQUIRE, __HIP_MEMORY_SCOPE_AGENT) < target)
      __builtin_amdgcn_s_sleep(1);
  }
  __syncthreads();
  __threadfence();
}

// ---------------------------------------------------------------------------
// Persistent sequential kernel. wg owns:
//  - stage A/D: row tiles wg*2, wg*2+1 (of 32)
//  - LSTM stages: within each of the 4 gate blocks, h-rows [wg*64, wg*64+64)
//    i.e. gate tiles tR = blk*64 + wg*4 + s,  s=0..3  (16 tiles, 2 per wave)
//  - stage E: wg<8 computes embed tile wg (Wm has 8 row tiles)
// ---------------------------------------------------------------------------
__global__ void __launch_bounds__(TPB, 1)
lstm_seq_k(const float* __restrict__ xin,
           const float* __restrict__ bs,
           const float* __restrict__ bih1, const float* __restrict__ bhh1,
           const float* __restrict__ bih2, const float* __restrict__ bhh2,
           const float* __restrict__ bl,  const float* __restrict__ bm,
           char* __restrict__ ws, float* __restrict__ out) {
  const int tid  = threadIdx.x;
  const int wg   = blockIdx.x;
  const int wave = tid >> 5;
  const int lane = tid & 31;

  unsigned* cnt = (unsigned*)(ws + OFF_CTRL);
  __bf16* inp = (__bf16*)(ws + OFF_INP);
  __bf16* h1b = (__bf16*)(ws + OFF_H1);
  __bf16* h2b = (__bf16*)(ws + OFF_H2);
  float*  c1  = (float*)(ws + OFF_C1);
  float*  c2  = (float*)(ws + OFF_C2);
  float*  yv  = (float*)(ws + OFF_Y);
  const v16bf* WsT   = (const v16bf*)(ws + OFF_WS_T);
  const v16bf* Wih1T = (const v16bf*)(ws + OFF_WIH1_T);
  const v16bf* Whh1T = (const v16bf*)(ws + OFF_WHH1_T);
  const v16bf* Wih2T = (const v16bf*)(ws + OFF_WIH2_T);
  const v16bf* Whh2T = (const v16bf*)(ws + OFF_WHH2_T);
  const v16bf* WlT   = (const v16bf*)(ws + OFF_WL_T);
  const v16bf* WmT   = (const v16bf*)(ws + OFF_WM_T);

  __shared__ __align__(32) __bf16 s_a[1024];
  __shared__ __align__(32) __bf16 s_b[1024];
  __shared__ float s_g[256];   // this wg's 256 gate rows
  __shared__ float s_r[256];   // reductions

  unsigned target = 0;

  for (int t = 0; t < T_STEPS; ++t) {
    const float* x = xin + (size_t)t * 360;
    const int pr = t & 1;
    __bf16* h1r = h1b + (pr ? 1024 : 0);
    __bf16* h1w = h1b + (pr ? 0 : 1024);
    __bf16* h2r = h2b + (pr ? 1024 : 0);
    __bf16* h2w = h2b + (pr ? 0 : 1024);

    // ---- Stage A: feat = Ws @ x + bs -> inp[0..511] (bf16) ----
    for (int i = tid; i < 384; i += TPB)
      s_a[i] = (__bf16)(i < 360 ? x[i] : 0.0f);
    __syncthreads();
    if (wave < 2) {
      int tR = wg * 2 + wave;
      v8f acc = {};
      acc = mv_tile(WsT, tR, 12, s_a, lane, acc);
      int r0 = tR * 16;
      if (lane == 0) {
#pragma unroll
        for (int r = 0; r < 8; ++r) inp[r0 + r] = (__bf16)(acc[r] + bs[r0 + r]);
      } else if (lane == 16) {
#pragma unroll
        for (int r = 0; r < 8; ++r) inp[r0 + 8 + r] = (__bf16)(acc[r] + bs[r0 + 8 + r]);
      }
    }
    gbarrier(cnt, target);

    // ---- Stage B: LSTM layer 1 (K = 640 inp + 1024 h1) ----
#if HAVE_ASYNC_LDS
    if (tid < 80)  async_cp16(inp + tid * 8, s_a + tid * 8);   // 1280B
    if (tid < 128) async_cp16(h1r + tid * 8, s_b + tid * 8);   // 2048B
    wait_async0();
#else
    for (int i = tid; i < 640;  i += TPB) s_a[i] = inp[i];
    for (int i = tid; i < 1024; i += TPB) s_b[i] = h1r[i];
#endif
    __syncthreads();
#pragma unroll
    for (int q = 0; q < 2; ++q) {
      int idx = wave * 2 + q;                 // 0..15
      int gb = idx >> 2, sb = idx & 3;
      int tR = gb * 64 + wg * 4 + sb;         // tile of 4096 gate rows
      v8f acc = {};
      acc = mv_tile(Wih1T, tR, 20, s_a, lane, acc);
      acc = mv_tile(Whh1T, tR, 32, s_b, lane, acc);
      int rg = tR * 16, rd = idx * 16;
      if (lane == 0) {
#pragma unroll
        for (int r = 0; r < 8; ++r) s_g[rd + r] = acc[r] + bih1[rg + r] + bhh1[rg + r];
      } else if (lane == 16) {
#pragma unroll
        for (int r = 0; r < 8; ++r) s_g[rd + 8 + r] = acc[r] + bih1[rg + 8 + r] + bhh1[rg + 8 + r];
      }
    }
    __syncthreads();
    if (tid < 64) {                            // h rows [wg*64, wg*64+64)
      int j = wg * 64 + tid;
      float gi = s_g[tid], gf = s_g[64 + tid], gg = s_g[128 + tid], go = s_g[192 + tid];
      float cn = sigmoidf_(gf) * c1[j] + sigmoidf_(gi) * tanhf(gg);
      c1[j] = cn;
      h1w[j] = (__bf16)(sigmoidf_(go) * tanhf(cn));
    }
    gbarrier(cnt, target);

    // ---- Stage C: LSTM layer 2 (K = 1024 h1_new + 1024 h2) ----
#if HAVE_ASYNC_LDS
    if (tid < 128) {
      async_cp16(h1w + tid * 8, s_a + tid * 8);
      async_cp16(h2r + tid * 8, s_b + tid * 8);
    }
    wait_async0();
#else
    for (int i = tid; i < 1024; i += TPB) { s_a[i] = h1w[i]; s_b[i] = h2r[i]; }
#endif
    __syncthreads();
#pragma unroll
    for (int q = 0; q < 2; ++q) {
      int idx = wave * 2 + q;
      int gb = idx >> 2, sb = idx & 3;
      int tR = gb * 64 + wg * 4 + sb;
      v8f acc = {};
      acc = mv_tile(Wih2T, tR, 32, s_a, lane, acc);
      acc = mv_tile(Whh2T, tR, 32, s_b, lane, acc);
      int rg = tR * 16, rd = idx * 16;
      if (lane == 0) {
#pragma unroll
        for (int r = 0; r < 8; ++r) s_g[rd + r] = acc[r] + bih2[rg + r] + bhh2[rg + r];
      } else if (lane == 16) {
#pragma unroll
        for (int r = 0; r < 8; ++r) s_g[rd + 8 + r] = acc[r] + bih2[rg + 8 + r] + bhh2[rg + 8 + r];
      }
    }
    __syncthreads();
    if (tid < 64) {
      int j = wg * 64 + tid;
      float gi = s_g[tid], gf = s_g[64 + tid], gg = s_g[128 + tid], go = s_g[192 + tid];
      float cn = sigmoidf_(gf) * c2[j] + sigmoidf_(gi) * tanhf(gg);
      c2[j] = cn;
      h2w[j] = (__bf16)(sigmoidf_(go) * tanhf(cn));
    }
    gbarrier(cnt, target);

    // ---- Stage D: y = Wl @ h2_new + bl ----
#if HAVE_ASYNC_LDS
    if (tid < 128) async_cp16(h2w + tid * 8, s_a + tid * 8);
    wait_async0();
#else
    for (int i = tid; i < 1024; i += TPB) s_a[i] = h2w[i];
#endif
    __syncthreads();
    if (wave < 2) {
      int tR = wg * 2 + wave;
      v8f acc = {};
      acc = mv_tile(WlT, tR, 32, s_a, lane, acc);
      int r0 = tR * 16;
      if (lane == 0) {
#pragma unroll
        for (int r = 0; r < 8; ++r) yv[r0 + r] = acc[r] + bl[r0 + r];
      } else if (lane == 16) {
#pragma unroll
        for (int r = 0; r < 8; ++r) yv[r0 + 8 + r] = acc[r] + bl[r0 + 8 + r];
      }
    }
    gbarrier(cnt, target);

    // ---- Stage E: log_softmax(y) -> out[t], embed = Wm @ lsm + bm ----
    float y0 = yv[tid], y1 = yv[tid + 256];
    s_r[tid] = fmaxf(y0, y1);
    __syncthreads();
    for (int o = 128; o > 0; o >>= 1) {
      if (tid < o) s_r[tid] = fmaxf(s_r[tid], s_r[tid + o]);
      __syncthreads();
    }
    float M = s_r[0];
    __syncthreads();
    s_r[tid] = __expf(y0 - M) + __expf(y1 - M);
    __syncthreads();
    for (int o = 128; o > 0; o >>= 1) {
      if (tid < o) s_r[tid] += s_r[tid + o];
      __syncthreads();
    }
    float L = __logf(s_r[0]);
    __syncthreads();
    s_a[tid]       = (__bf16)(y0 - M - L);
    s_a[tid + 256] = (__bf16)(y1 - M - L);
    if (tid < 32) out[(size_t)t * 512 + wg * 32 + tid] = yv[wg * 32 + tid] - M - L;
    __syncthreads();
    if (wg < 8 && wave == 0) {                // 8 embed tiles of 16 rows
      v8f acc = {};
      acc = mv_tile(WmT, wg, 16, s_a, lane, acc);
      int r0 = wg * 16;
      if (lane == 0) {
#pragma unroll
        for (int r = 0; r < 8; ++r) inp[512 + r0 + r] = (__bf16)(acc[r] + bm[r0 + r]);
      } else if (lane == 16) {
#pragma unroll
        for (int r = 0; r < 8; ++r) inp[512 + r0 + 8 + r] = (__bf16)(acc[r] + bm[r0 + 8 + r]);
      }
    }
    gbarrier(cnt, target);
  }
}

// ---------------------------------------------------------------------------
extern "C" void kernel_launch(void* const* d_in, const int* in_sizes, int n_in,
                              void* d_out, int out_size, void* d_ws, size_t ws_size,
                              hipStream_t stream) {
  (void)in_sizes; (void)n_in; (void)out_size; (void)ws_size;
  const float* inputVecs = (const float*)d_in[0];
  const float* Ws   = (const float*)d_in[1];
  const float* bs   = (const float*)d_in[2];
  const float* Wih1 = (const float*)d_in[3];
  const float* Whh1 = (const float*)d_in[4];
  const float* bih1 = (const float*)d_in[5];
  const float* bhh1 = (const float*)d_in[6];
  const float* Wih2 = (const float*)d_in[7];
  const float* Whh2 = (const float*)d_in[8];
  const float* bih2 = (const float*)d_in[9];
  const float* bhh2 = (const float*)d_in[10];
  const float* Wl   = (const float*)d_in[11];
  const float* bl   = (const float*)d_in[12];
  const float* Wm   = (const float*)d_in[13];
  const float* bm   = (const float*)d_in[14];
  char* ws = (char*)d_ws;
  float* out = (float*)d_out;

  // Weight tiling (runs every launch; deterministic).
  tile_weights_k<<<512,  TPB, 0, stream>>>(Ws,   (__bf16*)(ws + OFF_WS_T),   512,  360,  12);
  tile_weights_k<<<2048, TPB, 0, stream>>>(Wih1, (__bf16*)(ws + OFF_WIH1_T), 4096, 640,  20);
  tile_weights_k<<<2048, TPB, 0, stream>>>(Whh1, (__bf16*)(ws + OFF_WHH1_T), 4096, 1024, 32);
  tile_weights_k<<<2048, TPB, 0, stream>>>(Wih2, (__bf16*)(ws + OFF_WIH2_T), 4096, 1024, 32);
  tile_weights_k<<<2048, TPB, 0, stream>>>(Whh2, (__bf16*)(ws + OFF_WHH2_T), 4096, 1024, 32);
  tile_weights_k<<<512,  TPB, 0, stream>>>(Wl,   (__bf16*)(ws + OFF_WL_T),   512,  1024, 32);
  tile_weights_k<<<256,  TPB, 0, stream>>>(Wm,   (__bf16*)(ws + OFF_WM_T),   128,  512,  16);
  init_state_k<<<8, TPB, 0, stream>>>(ws);

  lstm_seq_k<<<NWG, TPB, 0, stream>>>(inputVecs, bs, bih1, bhh1, bih2, bhh2, bl, bm, ws, out);
}
